// AKT_58368605552760
// MI455X (gfx1250) — matrix operand order repaired and checked by reading, not compiled, not run on hardware
//
#include <hip/hip_runtime.h>
#include <math.h>

typedef __bf16 bf16;
typedef __bf16 bf16x16 __attribute__((ext_vector_type(16)));
typedef float  f32x8   __attribute__((ext_vector_type(8)));

#define Bsz   64
#define Ssz   256
#define Dsz   512
#define Hn    8
#define Fsz   2048
#define DHsz  64
#define Ln    6
#define FC1n  512
#define FC2n  256
#define NROWS (Bsz*Ssz)   // 16384

// ---------------------------------------------------------------------------
// CDNA5 async memory->LDS copy (ASYNCcnt-tracked, no VGPR round trip)
// ---------------------------------------------------------------------------
__device__ __forceinline__ void async_copy_b128(void* lds_dst, const void* gsrc) {
  unsigned off = (unsigned)(reinterpret_cast<unsigned long long>(lds_dst));
  asm volatile("global_load_async_to_lds_b128 %0, %1, off"
               :: "v"(off), "v"(gsrc) : "memory");
}
#define WAIT_ASYNC(n) asm volatile("s_wait_asynccnt " #n ::: "memory")

// ---------------------------------------------------------------------------
// Elementwise utility kernels
// ---------------------------------------------------------------------------
__global__ void cast_f32_to_bf16(const float* __restrict__ src, bf16* __restrict__ dst, long n) {
  long i = (long)blockIdx.x * blockDim.x + threadIdx.x;
  long st = (long)gridDim.x * blockDim.x;
  for (; i < n; i += st) dst[i] = (bf16)src[i];
}

__global__ void seed_state(const float* __restrict__ src, float* __restrict__ dstF,
                           bf16* __restrict__ dstB, long n) {
  long i = (long)blockIdx.x * blockDim.x + threadIdx.x;
  long st = (long)gridDim.x * blockDim.x;
  for (; i < n; i += st) { float v = src[i]; dstF[i] = v; dstB[i] = (bf16)v; }
}

__global__ void concat_cast(const float* __restrict__ xF, const float* __restrict__ qe,
                            bf16* __restrict__ dst) {
  long n = (long)NROWS * (2 * Dsz);
  long i = (long)blockIdx.x * blockDim.x + threadIdx.x;
  long st = (long)gridDim.x * blockDim.x;
  for (; i < n; i += st) {
    long r = i >> 10;           // row
    int  c = (int)(i & 1023);   // col in [0,1024)
    float v = (c < Dsz) ? xF[r * Dsz + c] : qe[r * Dsz + (c - Dsz)];
    dst[i] = (bf16)v;
  }
}

// ---------------------------------------------------------------------------
// bf16 WMMA GEMM:  C[M,N] = A[M,K] * W[K,N] + bias, optional ReLU.
// Block tile 64x128, K-step 64, 256 threads = 8 waves, each wave 16x64 of C.
// Double-buffered LDS staged with global_load_async_to_lds_b128; epilogue
// staged through LDS so all global stores are 128-bit.
// Exactly one of outF/outB must be non-null.
// ---------------------------------------------------------------------------
__global__ __launch_bounds__(256) void gemm_bf16(
    const bf16* __restrict__ A, const bf16* __restrict__ W,
    const float* __restrict__ bias,
    float* __restrict__ outF, bf16* __restrict__ outB,
    int M, int N, int K, int relu)
{
  __shared__ __align__(16) bf16 As[2][64][64];    // 16 KB
  __shared__ __align__(16) bf16 Bs[2][64][128];   // 32 KB (reused as C stage)

  const int tid  = threadIdx.x;
  const int lane = tid & 31;
  const int wave = tid >> 5;       // 0..7
  const int mt   = wave & 3;       // 16-row tile within 64
  const int ng   = wave >> 2;      // 0..1 -> 64-col group
  const int row0 = blockIdx.x * 64;
  const int col0 = blockIdx.y * 128;

  f32x8 acc[4];
  const f32x8 zero = {0.f,0.f,0.f,0.f,0.f,0.f,0.f,0.f};
  for (int j = 0; j < 4; ++j) acc[j] = zero;

  // issue one 64x64 A tile + 64x128 B tile as async LDS copies (6 per thread)
  auto issue = [&](int buf, int k0) {
#pragma unroll
    for (int t = 0; t < 2; ++t) {
      int id = tid + t * 256;        // 0..511
      int r = id >> 3, c = (id & 7) * 8;
      async_copy_b128(&As[buf][r][c], A + (size_t)(row0 + r) * K + k0 + c);
    }
#pragma unroll
    for (int t = 0; t < 4; ++t) {
      int id = tid + t * 256;        // 0..1023
      int r = id >> 4, c = (id & 15) * 8;
      async_copy_b128(&Bs[buf][r][c], W + (size_t)(k0 + r) * N + col0 + c);
    }
  };

  const int nk = K >> 6;
  issue(0, 0);
  for (int ks = 0; ks < nk; ++ks) {
    const int cur = ks & 1;
    if (ks + 1 < nk) {
      issue(1 - cur, (ks + 1) << 6);   // prefetch next tile into other buffer
      WAIT_ASYNC(6);                   // wait for *current* tile's 6 copies
    } else {
      WAIT_ASYNC(0);
    }
    __syncthreads();

#pragma unroll
    for (int kk = 0; kk < 2; ++kk) {
      bf16x16 afrag;
#pragma unroll
      for (int i = 0; i < 8; ++i) {
        int kb = ((i >> 2) << 4) + ((lane >> 4) << 3) + ((i & 3) << 1) + kk * 32;
        afrag[2*i]   = As[cur][mt*16 + (lane & 15)][kb];
        afrag[2*i+1] = As[cur][mt*16 + (lane & 15)][kb + 1];
      }
#pragma unroll
      for (int j = 0; j < 4; ++j) {
        bf16x16 bfrag;
        int n0 = ng * 64 + j * 16;
#pragma unroll
        for (int t = 0; t < 16; ++t) bfrag[t] = Bs[cur][kk * 32 + ((t >> 3) << 4) + ((lane >> 4) << 3) + (t & 7)][n0 + (lane & 15)];
        acc[j] = __builtin_amdgcn_wmma_f32_16x16x32_bf16(
            false, afrag, false, bfrag, (short)0, acc[j], false, false);
      }
    }
    __syncthreads();
  }

  // ---- epilogue: bias/ReLU in registers, stage C tile in LDS, b128 stores
  bf16*  cstB = (bf16*)&Bs[0][0][0];
  float* cstF = (float*)&Bs[0][0][0];
#pragma unroll
  for (int j = 0; j < 4; ++j) {
    int nl = ng * 64 + j * 16 + (lane & 15);
    float bv = bias ? bias[col0 + nl] : 0.0f;
#pragma unroll
    for (int v = 0; v < 8; ++v) {
      int ml = mt * 16 + v + ((lane >> 4) << 3);
      float r = acc[j][v] + bv;
      if (relu) r = fmaxf(r, 0.0f);
      if (outB) cstB[ml * 128 + nl] = (bf16)r;
      else      cstF[ml * 128 + nl] = r;
    }
  }
  __syncthreads();
  if (outB) {
#pragma unroll
    for (int t = 0; t < 4; ++t) {
      int id = tid + t * 256;          // 1024 x 8-bf16 chunks
      int r = id >> 4, c = (id & 15) * 8;
      *(uint4*)(outB + (size_t)(row0 + r) * N + col0 + c) =
          *(const uint4*)(cstB + r * 128 + c);
    }
  } else {
#pragma unroll
    for (int t = 0; t < 8; ++t) {
      int id = tid + t * 256;          // 2048 x 4-f32 chunks
      int r = id >> 5, c = (id & 31) * 4;
      *(uint4*)(outF + (size_t)(row0 + r) * N + col0 + c) =
          *(const uint4*)(cstF + r * 128 + c);
    }
  }
}

// ---------------------------------------------------------------------------
// Fused AKT attention for one (batch, head, 16-query tile).
// Q,K,V are bf16 [B,S,D] with head h in columns h*64..h*64+63.
// ---------------------------------------------------------------------------
__global__ __launch_bounds__(256) void akt_attention(
    const bf16* __restrict__ Qp, const bf16* __restrict__ Kp, const bf16* __restrict__ Vp,
    const float* __restrict__ gammas, int layer, int incl, int zero_pad,
    bf16* __restrict__ Ob)
{
  __shared__ __align__(16) bf16 Qt[16][DHsz];    //  2 KB (Q, then O stage)
  __shared__ __align__(16) bf16 KVt[Ssz][DHsz];  // 32 KB (K, then V)
  __shared__ float Sc[16][Ssz];                  // 16 KB
  __shared__ float redA[16][16];                 //  1 KB
  __shared__ float redB[16][16];                 //  1 KB

  const int tid  = threadIdx.x;
  const int lane = tid & 31;
  const int wave = tid >> 5;
  const int bh   = blockIdx.x;
  const int b    = bh / Hn, h = bh % Hn;
  const int q0   = blockIdx.y * 16;
  const size_t base = ((size_t)b * Ssz) * Dsz + (size_t)h * DHsz;

  // ---- async-stage Q tile (16x64) and K (256x64)
  if (tid < 128) {
    int r = tid >> 3, c = (tid & 7) * 8;
    async_copy_b128(&Qt[r][c], Qp + base + (size_t)(q0 + r) * Dsz + c);
  }
#pragma unroll
  for (int t = 0; t < 8; ++t) {
    int id = tid + t * 256;
    int r = id >> 3, c = (id & 7) * 8;
    async_copy_b128(&KVt[r][c], Kp + base + (size_t)r * Dsz + c);
  }
  WAIT_ASYNC(0);
  __syncthreads();

  // ---- scores = Q K^T / sqrt(DH); 16 N-tiles, 2 per wave, K=64 (2 WMMA each)
  const float scale = 0.125f;
#pragma unroll
  for (int jt = 0; jt < 2; ++jt) {
    int nt = wave * 2 + jt;
    f32x8 acc = {0.f,0.f,0.f,0.f,0.f,0.f,0.f,0.f};
#pragma unroll
    for (int ks = 0; ks < 2; ++ks) {
      int k0 = ks * 32;
      bf16x16 a, bfr;
#pragma unroll
      for (int i = 0; i < 8; ++i) {
        int kb = ((i >> 2) << 4) + ((lane >> 4) << 3) + ((i & 3) << 1) + k0;
        a[2*i]   = Qt[lane & 15][kb];
        a[2*i+1] = Qt[lane & 15][kb + 1];
      }
#pragma unroll
      for (int j = 0; j < 16; ++j) bfr[j] = KVt[nt * 16 + (lane & 15)][k0 + ((j >> 3) << 4) + ((lane >> 4) << 3) + (j & 7)];
      acc = __builtin_amdgcn_wmma_f32_16x16x32_bf16(
          false, a, false, bfr, (short)0, acc, false, false);
    }
#pragma unroll
    for (int v = 0; v < 8; ++v)
      Sc[v + ((lane >> 4) << 3)][nt * 16 + (lane & 15)] = acc[v] * scale;
  }
  __syncthreads();

  // ---- start V fetch NOW (async, overlaps the f32 row phase below)
#pragma unroll
  for (int t = 0; t < 8; ++t) {
    int id = tid + t * 256;
    int r = id >> 3, c = (id & 7) * 8;
    async_copy_b128(&KVt[r][c], Vp + base + (size_t)r * Dsz + c);
  }

  // ---- AKT distance-decay + double softmax: 16 threads per row
  {
    const int m = tid >> 4, sub = tid & 15;
    const int q = q0 + m;
    const int kb0 = sub * 16;
    const float NEG = -1e32f;
    const float gneg = -log1pf(expf(gammas[layer * Hn + h]));  // -softplus

    // 1: row max of masked scores
    float lmax = NEG;
#pragma unroll
    for (int k = kb0; k < kb0 + 16; ++k) {
      bool msk = incl ? (k <= q) : (k < q);
      lmax = fmaxf(lmax, msk ? Sc[m][k] : NEG);
    }
    redA[m][sub] = lmax;
    __syncthreads();
    float mx = NEG;
    for (int j = 0; j < 16; ++j) mx = fmaxf(mx, redA[m][j]);
    __syncthreads();

    // 2: softmax denominator (all) + masked numerator total
    float lsum = 0.f, ltot = 0.f;
#pragma unroll
    for (int k = kb0; k < kb0 + 16; ++k) {
      bool msk = incl ? (k <= q) : (k < q);
      float e = expf((msk ? Sc[m][k] : NEG) - mx);
      lsum += e;
      if (msk) ltot += e;
    }
    redA[m][sub] = lsum; redB[m][sub] = ltot;
    __syncthreads();
    float ssum = 0.f, smtot = 0.f;
    for (int j = 0; j < 16; ++j) { ssum += redA[m][j]; smtot += redB[m][j]; }
    const float inv = 1.0f / ssum;
    const float disttot = smtot * inv;
    // exclusive prefix of per-segment sm sums -> parallel inclusive cumsum
    float pref = 0.f;
    for (int j = 0; j < sub; ++j) pref += redB[m][j];
    pref *= inv;
    __syncthreads();

    // 3: distance decay, rescored logits, track new max
    float run = pref, lmx2 = NEG;
#pragma unroll
    for (int k = kb0; k < kb0 + 16; ++k) {
      bool msk = incl ? (k <= q) : (k < q);
      float s0 = Sc[m][k];
      float e = expf((msk ? s0 : NEG) - mx) * inv;
      run += msk ? e : 0.f;                        // inclusive cumsum of sm
      float pos = fabsf((float)(q - k));
      float dd = (disttot - run) * pos;
      dd = dd > 0.f ? dd : 0.f;
      float te = expf(sqrtf(dd) * gneg);
      te = fminf(fmaxf(te, 1e-5f), 1e5f);
      float s2 = msk ? s0 * te : NEG;
      Sc[m][k] = s2;
      lmx2 = fmaxf(lmx2, s2);
    }
    redA[m][sub] = lmx2;
    __syncthreads();
    float mx2 = NEG;
    for (int j = 0; j < 16; ++j) mx2 = fmaxf(mx2, redA[m][j]);
    float ls2 = 0.f;
#pragma unroll
    for (int k = kb0; k < kb0 + 16; ++k) ls2 += expf(Sc[m][k] - mx2);
    __syncthreads();
    redA[m][sub] = ls2;
    __syncthreads();
    float sum2 = 0.f;
    for (int j = 0; j < 16; ++j) sum2 += redA[m][j];
    const float inv2 = 1.0f / sum2;
#pragma unroll
    for (int k = kb0; k < kb0 + 16; ++k) {
      float a = expf(Sc[m][k] - mx2) * inv2;
      if (zero_pad && q == 0) a = 0.f;
      Sc[m][k] = a;                                // now holds attn weights
    }
  }
  WAIT_ASYNC(0);
  __syncthreads();

  // ---- O = attn @ V : [16,256]x[256,64]; 4 N-tiles on waves 0..3, K=256
  if (wave < 4) {
    int nt = wave;
    f32x8 acc = {0.f,0.f,0.f,0.f,0.f,0.f,0.f,0.f};
#pragma unroll
    for (int ks = 0; ks < 8; ++ks) {
      int k0 = ks * 32;
      bf16x16 a, bfr;
#pragma unroll
      for (int i = 0; i < 8; ++i) {
        int kb = ((i >> 2) << 4) + ((lane >> 4) << 3) + ((i & 3) << 1) + k0;
        a[2*i]   = (bf16)Sc[lane & 15][kb];
        a[2*i+1] = (bf16)Sc[lane & 15][kb + 1];
      }
#pragma unroll
      for (int j = 0; j < 16; ++j) bfr[j] = KVt[k0 + ((j >> 3) << 4) + ((lane >> 4) << 3) + (j & 7)][nt * 16 + (lane & 15)];
      acc = __builtin_amdgcn_wmma_f32_16x16x32_bf16(
          false, a, false, bfr, (short)0, acc, false, false);
    }
#pragma unroll
    for (int v = 0; v < 8; ++v) {                  // stage O tile in LDS (Qt)
      int m = v + ((lane >> 4) << 3);
      Qt[m][nt * 16 + (lane & 15)] = (bf16)acc[v];
    }
  }
  __syncthreads();
  if (tid < 128) {                                 // 128 x b128 coalesced stores
    int r = tid >> 3, c = (tid & 7) * 8;
    *(uint4*)(Ob + base + (size_t)(q0 + r) * Dsz + c) = *(const uint4*)(&Qt[r][c]);
  }
}

// ---------------------------------------------------------------------------
// q2 = LayerNorm(X + R) * g + b; writes f32 (may alias X) and bf16 copies.
// ---------------------------------------------------------------------------
__global__ __launch_bounds__(256) void ln_res(
    const float* __restrict__ X, const float* __restrict__ R,
    const float* __restrict__ g, const float* __restrict__ b,
    float* __restrict__ outF, bf16* __restrict__ outB)
{
  __shared__ float red[256];
  const int row = blockIdx.x, t = threadIdx.x;
  const size_t off = (size_t)row * Dsz;
  float v0 = X[off + t]       + R[off + t];
  float v1 = X[off + t + 256] + R[off + t + 256];
  red[t] = v0 + v1; __syncthreads();
  for (int o = 128; o > 0; o >>= 1) { if (t < o) red[t] += red[t + o]; __syncthreads(); }
  float mu = red[0] * (1.0f / Dsz); __syncthreads();
  float d0 = v0 - mu, d1 = v1 - mu;
  red[t] = d0 * d0 + d1 * d1; __syncthreads();
  for (int o = 128; o > 0; o >>= 1) { if (t < o) red[t] += red[t + o]; __syncthreads(); }
  float rstd = rsqrtf(red[0] * (1.0f / Dsz) + 1e-5f);
  float o0 = d0 * rstd * g[t]       + b[t];
  float o1 = d1 * rstd * g[t + 256] + b[t + 256];
  outF[off + t] = o0;        outF[off + t + 256] = o1;
  outB[off + t] = (bf16)o0;  outB[off + t + 256] = (bf16)o1;
}

// ---------------------------------------------------------------------------
// preds[row] = sigmoid(dot(h2[row,:256], O3) + bO3)
// ---------------------------------------------------------------------------
__global__ __launch_bounds__(256) void head_out(
    const bf16* __restrict__ h2, const float* __restrict__ O3,
    const float* __restrict__ bO3, float* __restrict__ out)
{
  __shared__ float red[256];
  const int row = blockIdx.x, t = threadIdx.x;
  red[t] = (float)h2[(size_t)row * FC2n + t] * O3[t];
  __syncthreads();
  for (int o = 128; o > 0; o >>= 1) { if (t < o) red[t] += red[t + o]; __syncthreads(); }
  if (t == 0) {
    float z = red[0] + bO3[0];
    out[row] = 1.0f / (1.0f + expf(-z));
  }
}

// ---------------------------------------------------------------------------
// Host orchestration
// ---------------------------------------------------------------------------
extern "C" void kernel_launch(void* const* d_in, const int* in_sizes, int n_in,
                              void* d_out, int out_size, void* d_ws, size_t ws_size,
                              hipStream_t stream) {
  (void)in_sizes; (void)n_in; (void)out_size; (void)ws_size;

  const float* q_embed = (const float*)d_in[0];
  const float* qa_embed= (const float*)d_in[1];
  const float* Wq = (const float*)d_in[2];  const float* bq = (const float*)d_in[3];
  const float* Wk = (const float*)d_in[4];  const float* bk = (const float*)d_in[5];
  const float* Wv = (const float*)d_in[6];  const float* bv = (const float*)d_in[7];
  const float* Wo = (const float*)d_in[8];  const float* bo = (const float*)d_in[9];
  const float* gammas = (const float*)d_in[10];
  const float* ln1g = (const float*)d_in[11]; const float* ln1b = (const float*)d_in[12];
  const float* W1 = (const float*)d_in[13]; const float* b1 = (const float*)d_in[14];
  const float* W2 = (const float*)d_in[15]; const float* b2 = (const float*)d_in[16];
  const float* ln2g = (const float*)d_in[17]; const float* ln2b = (const float*)d_in[18];
  const float* O1 = (const float*)d_in[19]; const float* bO1 = (const float*)d_in[20];
  const float* O2 = (const float*)d_in[21]; const float* bO2 = (const float*)d_in[22];
  const float* O3 = (const float*)d_in[23]; const float* bO3 = (const float*)d_in[24];

  // ---- workspace bump allocator (deterministic layout)
  char* w = (char*)d_ws;
  auto alloc = [&](size_t bytes) -> void* {
    void* p = (void*)w;
    w += (bytes + 255) & ~(size_t)255;
    return p;
  };
  const size_t nAct = (size_t)NROWS * Dsz;                 // 8,388,608
  bf16* wqB = (bf16*)alloc((size_t)Ln * Dsz * Dsz * 2);
  bf16* wkB = (bf16*)alloc((size_t)Ln * Dsz * Dsz * 2);
  bf16* wvB = (bf16*)alloc((size_t)Ln * Dsz * Dsz * 2);
  bf16* woB = (bf16*)alloc((size_t)Ln * Dsz * Dsz * 2);
  bf16* w1B = (bf16*)alloc((size_t)Ln * Dsz * Fsz * 2);
  bf16* w2B = (bf16*)alloc((size_t)Ln * Fsz * Dsz * 2);
  bf16* o1B = (bf16*)alloc((size_t)(2 * Dsz) * FC1n * 2);
  bf16* o2B = (bf16*)alloc((size_t)FC1n * FC2n * 2);
  float* xF = (float*)alloc(nAct * 4);
  float* yF = (float*)alloc(nAct * 4);
  bf16* xB = (bf16*)alloc(nAct * 2);
  bf16* yB = (bf16*)alloc(nAct * 2);
  bf16* qhB = (bf16*)alloc(nAct * 2);
  bf16* khB = (bf16*)alloc(nAct * 2);
  bf16* vhB = (bf16*)alloc(nAct * 2);
  bf16* aoB = (bf16*)alloc(nAct * 2);
  float* tmpF = (float*)alloc(nAct * 4);
  bf16* ffhB = (bf16*)alloc((size_t)NROWS * Fsz * 2);      // 67 MB
  // final stage aliases onto buffers no longer live
  bf16* catB = ffhB;                                       // needs 33.5 MB
  bf16* h1B  = qhB;                                        // 16.8 MB
  bf16* h2B  = khB;                                        // 8.4 MB

  // ---- weight conversion (f32 -> bf16)
  cast_f32_to_bf16<<<2048, 256, 0, stream>>>(Wq, wqB, (long)Ln * Dsz * Dsz);
  cast_f32_to_bf16<<<2048, 256, 0, stream>>>(Wk, wkB, (long)Ln * Dsz * Dsz);
  cast_f32_to_bf16<<<2048, 256, 0, stream>>>(Wv, wvB, (long)Ln * Dsz * Dsz);
  cast_f32_to_bf16<<<2048, 256, 0, stream>>>(Wo, woB, (long)Ln * Dsz * Dsz);
  cast_f32_to_bf16<<<4096, 256, 0, stream>>>(W1, w1B, (long)Ln * Dsz * Fsz);
  cast_f32_to_bf16<<<4096, 256, 0, stream>>>(W2, w2B, (long)Ln * Fsz * Dsz);
  cast_f32_to_bf16<<<1024, 256, 0, stream>>>(O1, o1B, (long)(2 * Dsz) * FC1n);
  cast_f32_to_bf16<<<512, 256, 0, stream>>>(O2, o2B, (long)FC1n * FC2n);

  // ---- running state: x from q_embed, y from qa_embed
  seed_state<<<4096, 256, 0, stream>>>(q_embed, xF, xB, (long)nAct);
  seed_state<<<4096, 256, 0, stream>>>(qa_embed, yF, yB, (long)nAct);

  struct Cfg { int qx; int vx; int incl; int zero; int pos; };
  const Cfg cfg[Ln] = {
    {0, 0, 1, 0, 1},   // blocks_1: y self-attn, inclusive, FFN
    {0, 0, 1, 0, 1},
    {1, 1, 1, 0, 0},   // blocks_2: x self-attn, inclusive, no FFN
    {1, 0, 0, 1, 1},   //           q=k=x, v=y, exclusive, zero_pad, FFN
    {1, 1, 1, 0, 0},
    {1, 0, 0, 1, 1},
  };

  dim3 blk(256);
  for (int i = 0; i < Ln; ++i) {
    float* qF = cfg[i].qx ? xF : yF;
    bf16*  qB = cfg[i].qx ? xB : yB;
    bf16*  vB = cfg[i].vx ? xB : yB;

    // Q/K/V projections (K input == Q input in all AKT blocks)
    dim3 gP(NROWS / 64, Dsz / 128);
    gemm_bf16<<<gP, blk, 0, stream>>>(qB, wqB + (size_t)i * Dsz * Dsz, bq + (size_t)i * Dsz,
                                      nullptr, qhB, NROWS, Dsz, Dsz, 0);
    gemm_bf16<<<gP, blk, 0, stream>>>(qB, wkB + (size_t)i * Dsz * Dsz, bk + (size_t)i * Dsz,
                                      nullptr, khB, NROWS, Dsz, Dsz, 0);
    gemm_bf16<<<gP, blk, 0, stream>>>(vB, wvB + (size_t)i * Dsz * Dsz, bv + (size_t)i * Dsz,
                                      nullptr, vhB, NROWS, Dsz, Dsz, 0);

    // fused AKT attention
    dim3 gA(Bsz * Hn, Ssz / 16);
    akt_attention<<<gA, blk, 0, stream>>>(qhB, khB, vhB, gammas, i,
                                          cfg[i].incl, cfg[i].zero, aoB);

    // output projection + residual layernorm (in place over the state)
    gemm_bf16<<<gP, blk, 0, stream>>>(aoB, woB + (size_t)i * Dsz * Dsz, bo + (size_t)i * Dsz,
                                      tmpF, nullptr, NROWS, Dsz, Dsz, 0);
    ln_res<<<NROWS, blk, 0, stream>>>(qF, tmpF, ln1g + (size_t)i * Dsz, ln1b + (size_t)i * Dsz,
                                      qF, qB);

    if (cfg[i].pos) {
      dim3 gF1(NROWS / 64, Fsz / 128);
      gemm_bf16<<<gF1, blk, 0, stream>>>(qB, w1B + (size_t)i * Dsz * Fsz, b1 + (size_t)i * Fsz,
                                         nullptr, ffhB, NROWS, Fsz, Dsz, 1);
      dim3 gF2(NROWS / 64, Dsz / 128);
      gemm_bf16<<<gF2, blk, 0, stream>>>(ffhB, w2B + (size_t)i * Fsz * Dsz, b2 + (size_t)i * Dsz,
                                         tmpF, nullptr, NROWS, Dsz, Fsz, 0);
      ln_res<<<NROWS, blk, 0, stream>>>(qF, tmpF, ln2g + (size_t)i * Dsz, ln2b + (size_t)i * Dsz,
                                        qF, qB);
    }
  }

  // ---- output head
  concat_cast<<<4096, blk, 0, stream>>>(xF, q_embed, catB);
  dim3 gO1(NROWS / 64, FC1n / 128);
  gemm_bf16<<<gO1, blk, 0, stream>>>(catB, o1B, bO1, nullptr, h1B, NROWS, FC1n, 2 * Dsz, 1);
  dim3 gO2(NROWS / 64, FC2n / 128);
  gemm_bf16<<<gO2, blk, 0, stream>>>(h1B, o2B, bO2, nullptr, h2B, NROWS, FC2n, FC1n, 1);
  head_out<<<NROWS, blk, 0, stream>>>(h2B, O3, bO3, (float*)d_out);
}
